// SwinTransformerBlock_65807488909318
// MI455X (gfx1250) — matrix operand order repaired and checked by reading, not compiled, not run on hardware
//
#include <hip/hip_runtime.h>
#include <hip/hip_bf16.h>

typedef __attribute__((ext_vector_type(8)))  _Float16 v8h;
typedef __attribute__((ext_vector_type(16))) _Float16 v16h;
typedef __attribute__((ext_vector_type(8)))  float    v8f;

#define DEVINL static __device__ __forceinline__

// ---------------------------------------------------------------------------
// WMMA helpers (CDNA5 v_wmma_f32_16x16x32_f16, wave32)
// ---------------------------------------------------------------------------
DEVINL v8f wmma_f16(v16h a, v16h b, v8f c) {
    return __builtin_amdgcn_wmma_f32_16x16x32_f16(
        /*neg_a=*/false, a, /*neg_b=*/false, b,
        /*c_mod=*/(short)0, c, /*reuse_a=*/false, /*reuse_b=*/false);
}

// A fragment (16x32 f16, MxK): lane 0-15 -> rows M=0..15, K = 8*h0 + e (e<8)
// and 16 + 8*h0 + (e-8) (e>=8), where h0 = lane/16. Two aligned 16B chunks.
DEVINL v16h load_a_frag(const _Float16* base, int ld, int row0, int k0, int lane) {
    const int m  = row0 + (lane & 15);
    const int h0 = (lane >> 4) & 1;
    const _Float16* p = base + (size_t)m * (size_t)ld + k0 + 8 * h0;
    v8h c0 = *(const v8h*)(p);
    v8h c1 = *(const v8h*)(p + 16);
    v16h r;
#pragma unroll
    for (int i = 0; i < 8; ++i) { r[i] = c0[i]; r[i + 8] = c1[i]; }
    return r;
}

// B fragment (32x16 f16, KxN) from a row-major weight W[n][k] (B = W^T):
// lane holds column n = lane&15; element e -> K = k0 + 8*h0 + e (e<8), k0 + 16 + 8*h0 + (e-8) (e>=8).
DEVINL v16h load_b_frag_rows(const _Float16* w, int ld, int n0, int k0, int lane) {
    const int n  = n0 + (lane & 15);
    const int h0 = (lane >> 4) & 1;
    const _Float16* p = w + (size_t)n * (size_t)ld + k0 + 8 * h0;
    v8h c0 = *(const v8h*)(p);
    v8h c1 = *(const v8h*)(p + 16);
    v16h r;
#pragma unroll
    for (int i = 0; i < 8; ++i) { r[i] = c0[i]; r[i + 8] = c1[i]; }
    return r;
}

// (window, token-in-window) -> linear index shifted by +2 in each 4D axis.
// Used both for the forward roll(-2) gather and the reverse roll(+2) scatter.
DEVINL int win_shift_linear(int wi, int ti) {
    int a = ((wi >> 6) & 3) * 4 + ((ti >> 6) & 3);
    int b = ((wi >> 4) & 3) * 4 + ((ti >> 4) & 3);
    int c = ((wi >> 2) & 3) * 4 + ((ti >> 2) & 3);
    int d = ((wi     ) & 3) * 4 + ((ti     ) & 3);
    a = (a + 2) & 15; b = (b + 2) & 15; c = (c + 2) & 15; d = (d + 2) & 15;
    return ((a * 16 + b) * 16 + c) * 16 + d;
}

// ---------------------------------------------------------------------------
// K0: f32 -> f16 weight conversion
// ---------------------------------------------------------------------------
__global__ void k_cvt_f16(const float* __restrict__ src, _Float16* __restrict__ dst, int n) {
    for (int i = blockIdx.x * blockDim.x + threadIdx.x; i < n; i += gridDim.x * blockDim.x)
        dst[i] = (_Float16)src[i];
}

// ---------------------------------------------------------------------------
// K1: LN1 + roll(-2,-2,-2,-2) + 4D window partition -> hwin f16 [win][ti][256]
// one wave per output token
// ---------------------------------------------------------------------------
__global__ void k_ln1_win(const float* __restrict__ x, const float* __restrict__ g,
                          const float* __restrict__ b, _Float16* __restrict__ hwin) {
    const int lane = threadIdx.x & 31;
    const int wid  = threadIdx.x >> 5;
    const int t    = blockIdx.x * 8 + wid;          // windowed token id
    const int wi = t >> 8, ti = t & 255;
    const int lsrc = win_shift_linear(wi, ti);
    const float* xr = x + (size_t)lsrc * 256;

    float vals[8], s = 0.f, ss = 0.f;
#pragma unroll
    for (int j = 0; j < 8; ++j) {
        float v = xr[lane + 32 * j];
        vals[j] = v; s += v; ss += v * v;
    }
#pragma unroll
    for (int m = 16; m >= 1; m >>= 1) {
        s  += __shfl_xor(s,  m, 32);
        ss += __shfl_xor(ss, m, 32);
    }
    const float mu   = s * (1.0f / 256.0f);
    const float var  = ss * (1.0f / 256.0f) - mu * mu;
    const float rstd = rsqrtf(var + 1e-5f);
#pragma unroll
    for (int j = 0; j < 8; ++j) {
        int c = lane + 32 * j;
        hwin[(size_t)t * 256 + c] = (_Float16)((vals[j] - mu) * rstd * g[c] + b[c]);
    }
}

// ---------------------------------------------------------------------------
// K2: QKV GEMM per window: (256x256) @ (256x768)^T; q scaled by 1/sqrt(32)
// 2x2 register blocking: one 32x32 output macro-tile per wave; 4 waves/block
// v is stored TRANSPOSED: vT[win][head][dim(32)][token(256)]
// ---------------------------------------------------------------------------
__global__ void k_qkv(const _Float16* __restrict__ hwin, const _Float16* __restrict__ wqkv,
                      const float* __restrict__ qkv_b,
                      _Float16* __restrict__ q, _Float16* __restrict__ k,
                      _Float16* __restrict__ vT) {
    const int lane = threadIdx.x & 31;
    const int tile = blockIdx.x * 4 + (threadIdx.x >> 5);   // 0 .. 49151
    const int win  = tile / 192;
    const int rem  = tile - win * 192;
    const int mt2  = rem / 24;            // 0..7   (rows/32)
    const int nt2  = rem - mt2 * 24;      // 0..23  (cols/32)

    const _Float16* A = hwin + (size_t)win * 65536;
    v8f acc00 = {}, acc01 = {}, acc10 = {}, acc11 = {};
#pragma unroll
    for (int kk = 0; kk < 8; ++kk) {
        const int k0 = kk * 32;
        v16h a0 = load_a_frag(A, 256, mt2 * 32,      k0, lane);
        v16h a1 = load_a_frag(A, 256, mt2 * 32 + 16, k0, lane);
        v16h b0 = load_b_frag_rows(wqkv, 256, nt2 * 32,      k0, lane);
        v16h b1 = load_b_frag_rows(wqkv, 256, nt2 * 32 + 16, k0, lane);
        acc00 = wmma_f16(a0, b0, acc00);
        acc01 = wmma_f16(a0, b1, acc01);
        acc10 = wmma_f16(a1, b0, acc10);
        acc11 = wmma_f16(a1, b1, acc11);
    }
    const int half = (lane >> 4) & 1;
    const int n_   = lane & 15;
    v8f accs[2][2] = {{acc00, acc01}, {acc10, acc11}};
#pragma unroll
    for (int im = 0; im < 2; ++im) {
#pragma unroll
        for (int in_ = 0; in_ < 2; ++in_) {
            const int j    = nt2 * 32 + in_ * 16 + n_;
            const int mat  = j >> 8;
            const int jh   = j & 255;
            const int head = jh >> 5;
            const int dd   = jh & 31;
            const float scale = (mat == 0) ? 0.17677669529663687f : 1.0f;
#pragma unroll
            for (int vv = 0; vv < 8; ++vv) {
                int ti = mt2 * 32 + im * 16 + vv + 8 * half;
                float val = (accs[im][in_][vv] + qkv_b[j]) * scale;
                if (mat == 0)
                    q[(((size_t)win * 8 + head) * 256 + ti) * 32 + dd] = (_Float16)val;
                else if (mat == 1)
                    k[(((size_t)win * 8 + head) * 256 + ti) * 32 + dd] = (_Float16)val;
                else
                    vT[(((size_t)win * 8 + head) * 32 + dd) * 256 + ti] = (_Float16)val;
            }
        }
    }
}

// ---------------------------------------------------------------------------
// K3: attention per (window, head): S = q@k^T + bias + mask, softmax, S@v
// 8 waves/block; each wave handles two 16-row blocks; P staged per-wave in LDS
// ---------------------------------------------------------------------------
__global__ void k_attn(const _Float16* __restrict__ q, const _Float16* __restrict__ k,
                       const _Float16* __restrict__ vT,
                       const float* __restrict__ bias_table,
                       const int* __restrict__ rel_index,
                       const float* __restrict__ attn_mask,
                       _Float16* __restrict__ attnw) {
    __shared__ __align__(16) _Float16 P[8 * 16 * 256];   // 64 KB, wave-private slots

    const int lane = threadIdx.x & 31;
    const int wid  = threadIdx.x >> 5;
    const int win  = blockIdx.x >> 3;
    const int head = blockIdx.x & 7;
    const int half = (lane >> 4) & 1;
    const int n_   = lane & 15;

    const _Float16* qb  = q  + ((size_t)win * 8 + head) * 256 * 32;
    const _Float16* kb  = k  + ((size_t)win * 8 + head) * 256 * 32;
    const _Float16* vTb = vT + ((size_t)win * 8 + head) * 32 * 256;
    _Float16* Pw = P + wid * 16 * 256;

    for (int pass = 0; pass < 2; ++pass) {
        const int mt = wid + pass * 8;
        v16h qa = load_a_frag(qb, 32, mt * 16, 0, lane);

        float sv[16][8];
#pragma unroll
        for (int nt = 0; nt < 16; ++nt) {
            v16h kf = load_b_frag_rows(kb, 32, nt * 16, 0, lane);
            v8f acc = {};
            acc = wmma_f16(qa, kf, acc);
            const int tj = nt * 16 + n_;
#pragma unroll
            for (int vv = 0; vv < 8; ++vv) {
                int ti = mt * 16 + vv + 8 * half;
                int rel = rel_index[ti * 256 + tj];
                float bm = bias_table[rel * 8 + head]
                         + attn_mask[(size_t)win * 65536 + (size_t)ti * 256 + tj];
                sv[nt][vv] = acc[vv] + bm;
            }
        }
        // row softmax (cross-lane over 16-lane halves)
#pragma unroll
        for (int vv = 0; vv < 8; ++vv) {
            float mx = -1e30f;
#pragma unroll
            for (int nt = 0; nt < 16; ++nt) mx = fmaxf(mx, sv[nt][vv]);
#pragma unroll
            for (int m = 8; m >= 1; m >>= 1) mx = fmaxf(mx, __shfl_xor(mx, m, 32));
            float sum = 0.f;
#pragma unroll
            for (int nt = 0; nt < 16; ++nt) {
                float e = __expf(sv[nt][vv] - mx);
                sv[nt][vv] = e; sum += e;
            }
#pragma unroll
            for (int m = 8; m >= 1; m >>= 1) sum += __shfl_xor(sum, m, 32);
            float inv = 1.0f / sum;
#pragma unroll
            for (int nt = 0; nt < 16; ++nt) sv[nt][vv] *= inv;
        }
        // write P row-block to this wave's LDS slot
#pragma unroll
        for (int nt = 0; nt < 16; ++nt)
#pragma unroll
            for (int vv = 0; vv < 8; ++vv)
                Pw[(vv + 8 * half) * 256 + nt * 16 + n_] = (_Float16)sv[nt][vv];
        asm volatile("s_wait_dscnt 0" ::: "memory");

        // PV: (16x256) @ (256x32); P fragment shared by both 16-wide dim tiles
        v8f acc0 = {}, acc1 = {};
#pragma unroll
        for (int kk = 0; kk < 8; ++kk) {
            const int k0 = kk * 32;
            v16h pa  = load_a_frag(Pw, 256, 0, k0, lane);
            v16h vf0 = load_b_frag_rows(vTb, 256, 0,  k0, lane);
            v16h vf1 = load_b_frag_rows(vTb, 256, 16, k0, lane);
            acc0 = wmma_f16(pa, vf0, acc0);
            acc1 = wmma_f16(pa, vf1, acc1);
        }
#pragma unroll
        for (int vv = 0; vv < 8; ++vv) {
            int ti = mt * 16 + vv + 8 * half;
            attnw[((size_t)win * 256 + ti) * 256 + head * 32 + n_]      = (_Float16)acc0[vv];
            attnw[((size_t)win * 256 + ti) * 256 + head * 32 + 16 + n_] = (_Float16)acc1[vv];
        }
    }
}

// ---------------------------------------------------------------------------
// K4: proj GEMM + window reverse + roll(+2) + residual -> y f32
// 2x2 register blocking: one 32x32 macro-tile per wave; 4 waves/block
// ---------------------------------------------------------------------------
__global__ void k_proj(const _Float16* __restrict__ attnw, const _Float16* __restrict__ wproj,
                       const float* __restrict__ proj_b, const float* __restrict__ x,
                       float* __restrict__ y) {
    const int lane = threadIdx.x & 31;
    const int tile = blockIdx.x * 4 + (threadIdx.x >> 5);   // 0 .. 16383
    const int win  = tile >> 6;
    const int mt2  = (tile >> 3) & 7;
    const int nt2  = tile & 7;

    const _Float16* A = attnw + (size_t)win * 65536;
    v8f acc00 = {}, acc01 = {}, acc10 = {}, acc11 = {};
#pragma unroll
    for (int kk = 0; kk < 8; ++kk) {
        const int k0 = kk * 32;
        v16h a0 = load_a_frag(A, 256, mt2 * 32,      k0, lane);
        v16h a1 = load_a_frag(A, 256, mt2 * 32 + 16, k0, lane);
        v16h b0 = load_b_frag_rows(wproj, 256, nt2 * 32,      k0, lane);
        v16h b1 = load_b_frag_rows(wproj, 256, nt2 * 32 + 16, k0, lane);
        acc00 = wmma_f16(a0, b0, acc00);
        acc01 = wmma_f16(a0, b1, acc01);
        acc10 = wmma_f16(a1, b0, acc10);
        acc11 = wmma_f16(a1, b1, acc11);
    }
    const int half = (lane >> 4) & 1;
    const int n_   = lane & 15;
    v8f accs[2][2] = {{acc00, acc01}, {acc10, acc11}};
#pragma unroll
    for (int im = 0; im < 2; ++im) {
#pragma unroll
        for (int in_ = 0; in_ < 2; ++in_) {
            const int c = nt2 * 32 + in_ * 16 + n_;
#pragma unroll
            for (int vv = 0; vv < 8; ++vv) {
                int ti   = mt2 * 32 + im * 16 + vv + 8 * half;
                int lfin = win_shift_linear(win, ti);
                y[(size_t)lfin * 256 + c] =
                    x[(size_t)lfin * 256 + c] + accs[im][in_][vv] + proj_b[c];
            }
        }
    }
}

// ---------------------------------------------------------------------------
// K5: LN2 over y -> h2 f16 (one wave per token)
// ---------------------------------------------------------------------------
__global__ void k_ln2(const float* __restrict__ y, const float* __restrict__ g,
                      const float* __restrict__ b, _Float16* __restrict__ h2) {
    const int lane = threadIdx.x & 31;
    const int t    = blockIdx.x * 8 + (threadIdx.x >> 5);
    const float* xr = y + (size_t)t * 256;
    float vals[8], s = 0.f, ss = 0.f;
#pragma unroll
    for (int j = 0; j < 8; ++j) {
        float v = xr[lane + 32 * j];
        vals[j] = v; s += v; ss += v * v;
    }
#pragma unroll
    for (int m = 16; m >= 1; m >>= 1) {
        s  += __shfl_xor(s,  m, 32);
        ss += __shfl_xor(ss, m, 32);
    }
    const float mu   = s * (1.0f / 256.0f);
    const float rstd = rsqrtf(ss * (1.0f / 256.0f) - mu * mu + 1e-5f);
#pragma unroll
    for (int j = 0; j < 8; ++j) {
        int c = lane + 32 * j;
        h2[(size_t)t * 256 + c] = (_Float16)((vals[j] - mu) * rstd * g[c] + b[c]);
    }
}

// ---------------------------------------------------------------------------
// K6: fused MLP per 32-token block: FC1 + GELU(exact) into LDS, FC2 + residual
// 8 waves/block; 2x2 register-blocked macro-tiles (32 rows x 32 cols)
// ---------------------------------------------------------------------------
__global__ void k_mlp(const _Float16* __restrict__ h2, const float* __restrict__ y,
                      const _Float16* __restrict__ wfc1, const float* __restrict__ fc1_b,
                      const _Float16* __restrict__ wfc2, const float* __restrict__ fc2_b,
                      float* __restrict__ out) {
    __shared__ __align__(16) _Float16 G[32 * 1024];   // 64 KB hidden activations

    const int lane = threadIdx.x & 31;
    const int wid  = threadIdx.x >> 5;
    const int half = (lane >> 4) & 1;
    const int n_   = lane & 15;
    const int t0   = blockIdx.x * 32;
    const _Float16* A0 = h2 + (size_t)t0 * 256;

    // stage 1: FC1 + GELU  (32 macro-tiles of 32x32 over N=1024; 4 per wave)
#pragma unroll 1
    for (int i = 0; i < 4; ++i) {
        const int nt2 = wid * 4 + i;                  // 0..31
        v8f acc00 = {}, acc01 = {}, acc10 = {}, acc11 = {};
#pragma unroll
        for (int kk = 0; kk < 8; ++kk) {
            const int k0 = kk * 32;
            v16h a0 = load_a_frag(A0, 256, 0,  k0, lane);
            v16h a1 = load_a_frag(A0, 256, 16, k0, lane);
            v16h b0 = load_b_frag_rows(wfc1, 256, nt2 * 32,      k0, lane);
            v16h b1 = load_b_frag_rows(wfc1, 256, nt2 * 32 + 16, k0, lane);
            acc00 = wmma_f16(a0, b0, acc00);
            acc01 = wmma_f16(a0, b1, acc01);
            acc10 = wmma_f16(a1, b0, acc10);
            acc11 = wmma_f16(a1, b1, acc11);
        }
        v8f accs[2][2] = {{acc00, acc01}, {acc10, acc11}};
#pragma unroll
        for (int im = 0; im < 2; ++im) {
#pragma unroll
            for (int in_ = 0; in_ < 2; ++in_) {
                const int cg = nt2 * 32 + in_ * 16 + n_;
#pragma unroll
                for (int vv = 0; vv < 8; ++vv) {
                    float xg = accs[im][in_][vv] + fc1_b[cg];
                    float gl = 0.5f * xg * (1.0f + erff(xg * 0.70710678118654752f));
                    G[(im * 16 + vv + 8 * half) * 1024 + cg] = (_Float16)gl;
                }
            }
        }
    }
    __syncthreads();

    // stage 2: FC2 + bias + residual (8 macro-tiles of 32x32 over N=256; 1/wave)
    {
        const int nt2 = wid;                          // 0..7
        v8f acc00 = {}, acc01 = {}, acc10 = {}, acc11 = {};
#pragma unroll 4
        for (int kk = 0; kk < 32; ++kk) {
            const int k0 = kk * 32;
            v16h a0 = load_a_frag(G, 1024, 0,  k0, lane);
            v16h a1 = load_a_frag(G, 1024, 16, k0, lane);
            v16h b0 = load_b_frag_rows(wfc2, 1024, nt2 * 32,      k0, lane);
            v16h b1 = load_b_frag_rows(wfc2, 1024, nt2 * 32 + 16, k0, lane);
            acc00 = wmma_f16(a0, b0, acc00);
            acc01 = wmma_f16(a0, b1, acc01);
            acc10 = wmma_f16(a1, b0, acc10);
            acc11 = wmma_f16(a1, b1, acc11);
        }
        v8f accs[2][2] = {{acc00, acc01}, {acc10, acc11}};
#pragma unroll
        for (int im = 0; im < 2; ++im) {
#pragma unroll
            for (int in_ = 0; in_ < 2; ++in_) {
                const int c = nt2 * 32 + in_ * 16 + n_;
#pragma unroll
                for (int vv = 0; vv < 8; ++vv) {
                    size_t row = (size_t)(t0 + im * 16 + vv + 8 * half);
                    out[row * 256 + c] = accs[im][in_][vv] + fc2_b[c] + y[row * 256 + c];
                }
            }
        }
    }
}

// ---------------------------------------------------------------------------
// host launcher
// ---------------------------------------------------------------------------
extern "C" void kernel_launch(void* const* d_in, const int* in_sizes, int n_in,
                              void* d_out, int out_size, void* d_ws, size_t ws_size,
                              hipStream_t stream) {
    const float* x          = (const float*)d_in[0];
    const float* g1         = (const float*)d_in[1];
    const float* b1         = (const float*)d_in[2];
    const float* qkv_w      = (const float*)d_in[3];
    const float* qkv_b      = (const float*)d_in[4];
    const float* bias_table = (const float*)d_in[5];
    const float* proj_w     = (const float*)d_in[6];
    const float* proj_b     = (const float*)d_in[7];
    const float* g2         = (const float*)d_in[8];
    const float* b2         = (const float*)d_in[9];
    const float* fc1_w      = (const float*)d_in[10];
    const float* fc1_b      = (const float*)d_in[11];
    const float* fc2_w      = (const float*)d_in[12];
    const float* fc2_b      = (const float*)d_in[13];
    const float* attn_mask  = (const float*)d_in[14];
    const int*   rel_index  = (const int*)d_in[15];
    float* out = (float*)d_out;

    char* ws = (char*)d_ws;
    const size_t MB = 1u << 20;
    _Float16* wqkv  = (_Float16*)(ws + 0);                    // 768*256  f16
    _Float16* wproj = (_Float16*)(ws + 393216);               // 256*256  f16
    _Float16* wfc1  = (_Float16*)(ws + 524288);               // 1024*256 f16
    _Float16* wfc2  = (_Float16*)(ws + 1048576);              // 256*1024 f16
    _Float16* hwin  = (_Float16*)(ws + 2 * MB);               // 32 MB
    _Float16* qb    = (_Float16*)(ws + 34 * MB);              // 32 MB
    _Float16* kb    = (_Float16*)(ws + 66 * MB);              // 32 MB
    _Float16* vTb   = (_Float16*)(ws + 98 * MB);              // 32 MB (transposed)
    _Float16* attnw = (_Float16*)(ws + 2 * MB);               // reuse hwin region
    float*    yb    = (float*)(ws + 34 * MB);                 // reuse q+k (64 MB)
    _Float16* h2    = (_Float16*)(ws + 98 * MB);              // reuse vT region

    k_cvt_f16<<<768, 256, 0, stream>>>(qkv_w, wqkv, 768 * 256);
    k_cvt_f16<<<256, 256, 0, stream>>>(proj_w, wproj, 256 * 256);
    k_cvt_f16<<<1024, 256, 0, stream>>>(fc1_w, wfc1, 1024 * 256);
    k_cvt_f16<<<1024, 256, 0, stream>>>(fc2_w, wfc2, 256 * 1024);

    k_ln1_win<<<8192, 256, 0, stream>>>(x, g1, b1, hwin);
    k_qkv<<<12288, 128, 0, stream>>>(hwin, wqkv, qkv_b, qb, kb, vTb);
    k_attn<<<2048, 256, 0, stream>>>(qb, kb, vTb, bias_table, rel_index, attn_mask, attnw);
    k_proj<<<4096, 128, 0, stream>>>(attnw, wproj, proj_b, x, yb);
    k_ln2<<<8192, 256, 0, stream>>>(yb, g2, b2, h2);
    k_mlp<<<2048, 256, 0, stream>>>(h2, yb, wfc1, fc1_b, wfc2, fc2_b, out);
}
